// AttentionTCCNet_28587302322733
// MI455X (gfx1250) — matrix operand order repaired and checked by reading, not compiled, not run on hardware
//
#include <hip/hip_runtime.h>
#include <hip/hip_bf16.h>
#include <math.h>

typedef __attribute__((ext_vector_type(16))) _Float16 v16h;
typedef __attribute__((ext_vector_type(2)))  _Float16 v2h;
typedef __attribute__((ext_vector_type(8)))  float    v8f;

__device__ __forceinline__ float sigf(float x) { return 1.0f / (1.0f + __expf(-x)); }

// ---------------- K0: raw = rgb*conf, block min/max partials ----------------
__global__ void k_raw_minmax(const float* __restrict__ rgb, const float* __restrict__ conf,
                             float* __restrict__ raw, float* __restrict__ pmin,
                             float* __restrict__ pmax) {
  __shared__ float smn[256], smx[256];
  int idx = blockIdx.x * 256 + threadIdx.x;          // exactly 196608 total
  int t = idx / 12288;
  int pix = idx & 4095;
  float v = rgb[idx] * conf[t * 4096 + pix];
  raw[idx] = v;
  smn[threadIdx.x] = v; smx[threadIdx.x] = v;
  __syncthreads();
  for (int s = 128; s > 0; s >>= 1) {
    if (threadIdx.x < s) {
      smn[threadIdx.x] = fminf(smn[threadIdx.x], smn[threadIdx.x + s]);
      smx[threadIdx.x] = fmaxf(smx[threadIdx.x], smx[threadIdx.x + s]);
    }
    __syncthreads();
  }
  if (threadIdx.x == 0) { pmin[blockIdx.x] = smn[0]; pmax[blockIdx.x] = smx[0]; }
}

// ---------------- K1: finalize global min/max ----------------
__global__ void k_minmax_final(const float* __restrict__ pmin, const float* __restrict__ pmax,
                               float* __restrict__ mm) {
  __shared__ float smn[256], smx[256];
  int t = threadIdx.x;
  float mn = 1e30f, mx = -1e30f;
  for (int i = t; i < 768; i += 256) { mn = fminf(mn, pmin[i]); mx = fmaxf(mx, pmax[i]); }
  smn[t] = mn; smx[t] = mx;
  __syncthreads();
  for (int s = 128; s > 0; s >>= 1) {
    if (t < s) { smn[t] = fminf(smn[t], smn[t + s]); smx[t] = fmaxf(smx[t], smx[t + s]); }
    __syncthreads();
  }
  if (t == 0) { mm[0] = smn[0]; mm[1] = smx[0]; mm[2] = 1.0f / (smx[0] - smn[0]); }
}

// ---------------- K2: s = (raw-mn)*inv, per-(t,c) spatial sums ----------------
__global__ void k_scale_sums(const float* __restrict__ raw, const float* __restrict__ mm,
                             float* __restrict__ s, float* __restrict__ csum) {
  __shared__ float red[256];
  int b = blockIdx.x;            // t*3+c, 48 blocks
  int base = b * 4096;
  float mn = mm[0], inv = mm[2];
  float acc = 0.f;
  for (int i = threadIdx.x; i < 4096; i += 256) {
    float v = (raw[base + i] - mn) * inv;
    s[base + i] = v;
    acc += v;
  }
  red[threadIdx.x] = acc;
  __syncthreads();
  for (int st = 128; st > 0; st >>= 1) {
    if (threadIdx.x < st) red[threadIdx.x] += red[threadIdx.x + st];
    __syncthreads();
  }
  if (threadIdx.x == 0) csum[b] = red[0];
}

// ---------------- K3: wts = softmax(att_x)/T  (att_h cancels) ----------------
__global__ void k_attn_wts(const float* __restrict__ csum, const float* __restrict__ phix,
                           float* __restrict__ wtsn) {
  if (threadIdx.x != 0) return;
  float att[16];
  float mx = -1e30f;
  for (int t = 0; t < 16; ++t) {
    float a = (csum[t * 3 + 0] * phix[0] + csum[t * 3 + 1] * phix[1] +
               csum[t * 3 + 2] * phix[2]) * (1.0f / 4096.0f);
    att[t] = a; mx = fmaxf(mx, a);
  }
  float sum = 0.f;
  for (int t = 0; t < 16; ++t) { att[t] = __expf(att[t] - mx); sum += att[t]; }
  float r = 1.0f / (sum * 16.0f);
  for (int t = 0; t < 16; ++t) wtsn[t] = att[t] * r;
}

// ---------------- K4: x = sum_t s[t]*wts_n[t]  -> [3,4096] ----------------
__global__ void k_attn_x(const float* __restrict__ s, const float* __restrict__ wtsn,
                         float* __restrict__ xbuf) {
  int idx = blockIdx.x * 256 + threadIdx.x;  // 12288 = c*4096+pix
  float acc = 0.f;
  for (int t = 0; t < 16; ++t) acc += s[t * 12288 + idx] * wtsn[t];
  xbuf[idx] = acc;
}

// ------- K5: gates_x[pix][m'] = bias + conv(x, lstm_w[:, 0:3]) (pixel-major, gate-interleaved) -------
__global__ void k_gates_x(const float* __restrict__ lstm_w, const float* __restrict__ lstm_b,
                          const float* __restrict__ xbuf, float* __restrict__ gx) {
  int gid = blockIdx.x * 256 + threadIdx.x;  // 4096*512, gid = pix*512 + m'
  int pix = gid >> 9;
  int mp = gid & 511;                        // interleaved row m' = 4c+gate
  int m = (mp & 3) * 128 + (mp >> 2);        // original row
  int y = pix >> 6, x = pix & 63;
  float acc = lstm_b[m];
  for (int ci = 0; ci < 3; ++ci) {
    const float* wp = lstm_w + (m * 131 + ci) * 25;
    const float* ip = xbuf + ci * 4096;
    for (int ky = 0; ky < 5; ++ky) {
      int ys = y + ky - 2;
      if (ys < 0 || ys >= 64) continue;
      for (int kx = 0; kx < 5; ++kx) {
        int xs = x + kx - 2;
        if (xs < 0 || xs >= 64) continue;
        acc += wp[ky * 5 + kx] * ip[ys * 64 + xs];
      }
    }
  }
  gx[gid] = acc;
}

// ------- K6: pack lstm_w[:,3:131] into WMMA A-fragment order (f16) -------
// layout: [kIter(100)][mTile(32)][lane(32)][e(16)], kIter = kb*25+tap, ci = kb*32+kk
__global__ void k_pack_w(const float* __restrict__ lstm_w, _Float16* __restrict__ af) {
  int gid = blockIdx.x * 256 + threadIdx.x;  // 1,638,400 exact
  int e    = gid & 15;
  int lane = (gid >> 4) & 31;
  int mt   = (gid >> 9) & 31;
  int kIter = gid >> 14;                     // 0..99
  int kb = kIter / 25, tap = kIter - kb * 25;
  // contiguous per-lane K mapping (k = 16*(lane/16) + e), matches B staging
  int kk = ((lane < 16) ? 0 : 16) + e;
  int ci = kb * 32 + kk;
  int mp = mt * 16 + (lane & 15);            // interleaved row
  int m = (mp & 3) * 128 + (mp >> 2);
  af[gid] = (_Float16)lstm_w[(m * 131 + 3 + ci) * 25 + tap];
}

// ---------------- K7: zero state ----------------
__global__ void k_init_state(float* __restrict__ cbuf, float* __restrict__ hsum,
                             _Float16* __restrict__ h16a, _Float16* __restrict__ h16b) {
  int i = blockIdx.x * 256 + threadIdx.x;    // 524288
  cbuf[i] = 0.f; hsum[i] = 0.f;
  h16a[i] = (_Float16)0.f; h16b[i] = (_Float16)0.f;
}

// ------- K8: one ConvLSTM step: WMMA implicit-GEMM + fused gate update -------
// 128 threads = 4 waves; each wave: M=64 (4 tiles) x N=32 (2 tiles) -> 8 WMMA per tap.
// grid (32, 8): blockIdx.x = N block (128 pixels = 2 image rows), blockIdx.y = M block (64 rows).
// All state tensors pixel-major: h16[pix][128], cbuf/hsum[pix][128], gx[pix][512]
__global__ void __launch_bounds__(128, 2)
k_convlstm_step(const _Float16* __restrict__ af, const _Float16* __restrict__ hin,
                const float* __restrict__ gx, float* __restrict__ cbuf,
                float* __restrict__ hsum, _Float16* __restrict__ hout) {
  // LDS tile: (6 halo rows x 68 cols) x 32 channels, channel-contiguous
  __shared__ __align__(32) _Float16 hs[6 * 68 * 32];   // 26112 B
  const int tid  = threadIdx.x;
  const int lane = tid & 31;
  const int wv   = tid >> 5;                 // 4 waves = N strips
  const int ylocal = wv >> 1;                // which of 2 image rows
  const int x0     = (wv & 1) * 32;          // 32-wide strip in the row
  const int yrow0  = blockIdx.x * 2;
  const int m_base  = blockIdx.y * 64;
  const int mTileG0 = m_base >> 4;
  const int ln    = lane & 15;
  const int Kbase = (lane < 16) ? 0 : 16;    // contiguous B per-lane K half: k = 16*(lane/16) + e
  const int mOff  = (lane < 16) ? 0 : 8;     // ISA 32-bit C 16x16 per-lane M half
  const int pix0  = (yrow0 + ylocal) * 64 + x0 + ln;   // N tile 0
  const int pix1  = pix0 + 16;                          // N tile 1

  // accumulators seeded from gates_x: 8 consecutive rows -> one v8f load per tile
  v8f a0, a1, a2, a3, b0, b1, b2, b3;
  {
    const float* g0 = gx + (size_t)pix0 * 512 + m_base + mOff;
    const float* g1 = gx + (size_t)pix1 * 512 + m_base + mOff;
    a0 = *(const v8f*)(g0 +  0); a1 = *(const v8f*)(g0 + 16);
    a2 = *(const v8f*)(g0 + 32); a3 = *(const v8f*)(g0 + 48);
    b0 = *(const v8f*)(g1 +  0); b1 = *(const v8f*)(g1 + 16);
    b2 = *(const v8f*)(g1 + 32); b3 = *(const v8f*)(g1 + 48);
  }

#pragma unroll 1
  for (int kb = 0; kb < 4; ++kb) {
    __syncthreads();
    // stage 32 channels with 6x68 halo; each (ry,cx) is 64B contiguous in global
#pragma unroll 1
    for (int idx = tid; idx < 408 * 4; idx += 128) {
      int pos  = idx >> 2;                   // 0..407
      int part = idx & 3;                    // 16B sub-chunk
      int ry = pos / 68;
      int cx = pos - ry * 68;
      int ys = yrow0 + ry - 2;
      int xs = cx - 2;
      uint4 v = make_uint4(0u, 0u, 0u, 0u);
      if (ys >= 0 && ys < 64 && xs >= 0 && xs < 64)
        v = *(const uint4*)(hin + (size_t)(ys * 64 + xs) * 128 + kb * 32 + part * 8);
      *(uint4*)(&hs[pos * 32 + part * 8]) = v;
    }
    __syncthreads();
    // keep rolled: one load clause + one wait + pure 8-WMMA burst per tap,
    // accumulators pinned in place (no inter-iteration register rotation)
#pragma unroll 1
    for (int tap = 0; tap < 25; ++tap) {
      const int kIter = kb * 25 + tap;
      const int ry  = tap / 5;
      const int dxr = tap - ry * 5;
      const _Float16* bp = &hs[((ylocal + ry) * 68 + x0 + ln + dxr) * 32 + Kbase];
      v16h fb0 = *(const v16h*)(bp);
      v16h fb1 = *(const v16h*)(bp + 512);   // +16 columns
      const _Float16* ap = af + (size_t)(kIter * 32 + mTileG0) * 512 + lane * 16;
      v16h fa0 = *(const v16h*)(ap);
      v16h fa1 = *(const v16h*)(ap + 512);
      v16h fa2 = *(const v16h*)(ap + 1024);
      v16h fa3 = *(const v16h*)(ap + 1536);
      a0 = __builtin_amdgcn_wmma_f32_16x16x32_f16(false, fa0, false, fb0, (short)0, a0, false, false);
      a1 = __builtin_amdgcn_wmma_f32_16x16x32_f16(false, fa1, false, fb0, (short)0, a1, false, false);
      a2 = __builtin_amdgcn_wmma_f32_16x16x32_f16(false, fa2, false, fb0, (short)0, a2, false, false);
      a3 = __builtin_amdgcn_wmma_f32_16x16x32_f16(false, fa3, false, fb0, (short)0, a3, false, false);
      b0 = __builtin_amdgcn_wmma_f32_16x16x32_f16(false, fa0, false, fb1, (short)0, b0, false, false);
      b1 = __builtin_amdgcn_wmma_f32_16x16x32_f16(false, fa1, false, fb1, (short)0, b1, false, false);
      b2 = __builtin_amdgcn_wmma_f32_16x16x32_f16(false, fa2, false, fb1, (short)0, b2, false, false);
      b3 = __builtin_amdgcn_wmma_f32_16x16x32_f16(false, fa3, false, fb1, (short)0, b3, false, false);
    }
  }

  // fused LSTM pointwise update: each lane holds i/f/o/g for 2 adjacent channels per tile
  v8f accs[8] = {a0, a1, a2, a3, b0, b1, b2, b3};
#pragma unroll
  for (int n = 0; n < 2; ++n) {
    const int pix = n ? pix1 : pix0;
#pragma unroll
    for (int t = 0; t < 4; ++t) {
      v8f acc = accs[n * 4 + t];
      const int ch0 = (m_base + t * 16 + mOff) >> 2;   // even; pair = (ch0, ch0+1)
      const size_t base = (size_t)pix * 128 + ch0;
      float2 cold = *(const float2*)(cbuf + base);
      float cn0 = sigf(acc[1]) * cold.x + sigf(acc[0]) * tanhf(acc[3]);
      float cn1 = sigf(acc[5]) * cold.y + sigf(acc[4]) * tanhf(acc[7]);
      float hn0 = sigf(acc[2]) * tanhf(cn0);
      float hn1 = sigf(acc[6]) * tanhf(cn1);
      *(float2*)(cbuf + base) = make_float2(cn0, cn1);
      v2h hh; hh[0] = (_Float16)hn0; hh[1] = (_Float16)hn1;
      *(v2h*)(hout + base) = hh;
      float2 hsold = *(const float2*)(hsum + base);
      *(float2*)(hsum + base) = make_float2(hsold.x + hn0, hsold.y + hn1);
    }
  }
}

// ---------------- K9: hmean + maxpool 3x3 s2 ceil -> [128,32,32] (hsum pixel-major) ----------------
__global__ void k_pool(const float* __restrict__ hsum, float* __restrict__ pool) {
  int gid = blockIdx.x * 256 + threadIdx.x;  // 131072
  int ch = gid >> 10;
  int r = gid & 1023;
  int py = r >> 5, px = r & 31;
  float m = -1e30f;
  for (int wy = 0; wy < 3; ++wy) {
    int y = 2 * py + wy; if (y >= 64) break;
    for (int wx = 0; wx < 3; ++wx) {
      int x = 2 * px + wx; if (x >= 64) break;
      m = fmaxf(m, hsum[(size_t)(y * 64 + x) * 128 + ch]);
    }
  }
  pool[gid] = m * (1.0f / 16.0f);
}

// ---------------- K10: conv1 6x6 pad3 + sigmoid -> [128,33,33] ----------------
__global__ void k_conv1(const float* __restrict__ pool, const float* __restrict__ w,
                        const float* __restrict__ b, float* __restrict__ y1) {
  int gid = blockIdx.x * 256 + threadIdx.x;
  if (gid >= 128 * 1089) return;
  int co = gid / 1089;
  int r = gid - co * 1089;
  int oy = r / 33;
  int ox = r - oy * 33;
  float acc = b[co];
  for (int ci = 0; ci < 128; ++ci) {
    const float* wp = w + (co * 128 + ci) * 36;
    const float* ip = pool + ci * 1024;
    for (int ky = 0; ky < 6; ++ky) {
      int ys = oy - 3 + ky;
      if (ys < 0 || ys >= 32) continue;
      for (int kx = 0; kx < 6; ++kx) {
        int xs = ox - 3 + kx;
        if (xs < 0 || xs >= 32) continue;
        acc += wp[ky * 6 + kx] * ip[ys * 32 + xs];
      }
    }
  }
  y1[gid] = sigf(acc);
}

// ---------------- K11: conv2 1x1 + sigmoid + spatial sum -> v[3] ----------------
__global__ void k_conv2_reduce(const float* __restrict__ y1, const float* __restrict__ w,
                               const float* __restrict__ b, float* __restrict__ vv) {
  __shared__ float red[256];
  int c = blockIdx.x;
  float acc = 0.f;
  for (int pix = threadIdx.x; pix < 1089; pix += 256) {
    float dot = b[c];
    for (int ci = 0; ci < 128; ++ci)
      dot += w[c * 128 + ci] * y1[ci * 1089 + pix];
    acc += sigf(dot);
  }
  red[threadIdx.x] = acc;
  __syncthreads();
  for (int s = 128; s > 0; s >>= 1) {
    if (threadIdx.x < s) red[threadIdx.x] += red[threadIdx.x + s];
    __syncthreads();
  }
  if (threadIdx.x == 0) vv[c] = red[0];
}

// ---------------- K12: L2 normalize -> pred[3] ----------------
__global__ void k_norm(const float* __restrict__ vv, float* __restrict__ out) {
  if (threadIdx.x == 0 && blockIdx.x == 0) {
    float a = vv[0], b = vv[1], c = vv[2];
    float n = sqrtf(a * a + b * b + c * c);
    n = fmaxf(n, 1e-12f);
    out[0] = a / n; out[1] = b / n; out[2] = c / n;
  }
}

extern "C" void kernel_launch(void* const* d_in, const int* in_sizes, int n_in,
                              void* d_out, int out_size, void* d_ws, size_t ws_size,
                              hipStream_t stream) {
  const float* rgb   = (const float*)d_in[0];
  const float* conf  = (const float*)d_in[1];
  const float* phix  = (const float*)d_in[2];
  const float* phih  = (const float*)d_in[3];  // mathematically cancels in softmax
  const float* lstmw = (const float*)d_in[4];
  const float* lstmb = (const float*)d_in[5];
  const float* c1w   = (const float*)d_in[6];
  const float* c1b   = (const float*)d_in[7];
  const float* c2w   = (const float*)d_in[8];
  const float* c2b   = (const float*)d_in[9];
  float* out = (float*)d_out;
  (void)phih; (void)in_sizes; (void)n_in; (void)out_size; (void)ws_size;

  float* ws   = (float*)d_ws;
  float* RAW  = ws;                  // 196608
  float* S    = RAW + 196608;        // 196608
  float* PMIN = S + 196608;          // 768
  float* PMAX = PMIN + 768;          // 768
  float* MM   = PMAX + 768;          // 8
  float* CSUM = MM + 8;              // 48
  float* WTSN = CSUM + 48;           // 16
  float* XBUF = WTSN + 16;           // 12288
  float* GX   = XBUF + 12288;        // 2097152 (32B-aligned offset)
  float* CBUF = GX + 2097152;        // 524288
  float* HSUM = CBUF + 524288;       // 524288
  float* POOL = HSUM + 524288;       // 131072
  float* Y1   = POOL + 131072;       // 139392
  float* VV   = Y1 + 139392;         // 16
  _Float16* H16A = (_Float16*)(VV + 16);   // 524288 halves (16B-aligned offset)
  _Float16* H16B = H16A + 524288;          // 524288 halves
  _Float16* AF   = H16B + 524288;          // 1,638,400 halves (packed weights, 32B-aligned)

  k_raw_minmax<<<768, 256, 0, stream>>>(rgb, conf, RAW, PMIN, PMAX);
  k_minmax_final<<<1, 256, 0, stream>>>(PMIN, PMAX, MM);
  k_scale_sums<<<48, 256, 0, stream>>>(RAW, MM, S, CSUM);
  k_attn_wts<<<1, 32, 0, stream>>>(CSUM, phix, WTSN);
  k_attn_x<<<48, 256, 0, stream>>>(S, WTSN, XBUF);
  k_gates_x<<<8192, 256, 0, stream>>>(lstmw, lstmb, XBUF, GX);
  k_pack_w<<<6400, 256, 0, stream>>>(lstmw, AF);
  k_init_state<<<2048, 256, 0, stream>>>(CBUF, HSUM, H16A, H16B);

  for (int t = 0; t < 16; ++t) {
    const _Float16* hin = (t & 1) ? H16B : H16A;
    _Float16* hout = (t & 1) ? H16A : H16B;
    k_convlstm_step<<<dim3(32, 8), 128, 0, stream>>>(AF, hin, GX, CBUF, HSUM, hout);
  }

  k_pool<<<512, 256, 0, stream>>>(HSUM, POOL);
  k_conv1<<<545, 256, 0, stream>>>(POOL, c1w, c1b, Y1);
  k_conv2_reduce<<<3, 256, 0, stream>>>(Y1, c2w, c2b, VV);
  k_norm<<<1, 32, 0, stream>>>(VV, out);
}
